// GatedAttentionUnit_29841432773266
// MI455X (gfx1250) — matrix operand order repaired and checked by reading, not compiled, not run on hardware
//
#include <hip/hip_runtime.h>

// ---------------------------------------------------------------------------
// GatedAttentionUnit on MI455X (gfx1250)
//  - all GEMMs via v_wmma_f32_16x16x32_bf16 (f32 accumulate)
//  - HBM -> LDS staging via Tensor Data Mover (tensor_load_to_lds),
//    double buffered, s_wait_tensorcnt synchronized
//  - B matrices pre-transposed so both WMMA operands are contiguous-K rows
//  - sched_barrier pins loads-before-WMMAs: one dscnt wait, 8 WMMAs back-to-back
// ---------------------------------------------------------------------------

#define HID  512
#define PROJ 1024
#define ATTN 128
#define BB   4
#define LL   2048
#define NTOK (BB * LL)         // 8192
#define NW1  (2 * PROJ + ATTN) // 2176

#define BM 64
#define BN 256
#define BK 32

typedef __attribute__((ext_vector_type(16))) __bf16 v16bf;
typedef __attribute__((ext_vector_type(8)))  float  v8f;
typedef unsigned int u32x4 __attribute__((ext_vector_type(4)));
typedef int          i32x4 __attribute__((ext_vector_type(4)));
typedef int          i32x8 __attribute__((ext_vector_type(8)));

union V16U { v16bf v; uint4 q[2]; };

#if defined(__HIP_DEVICE_COMPILE__) && __has_builtin(__builtin_amdgcn_tensor_load_to_lds)
#define HAVE_TDM 1
#else
#define HAVE_TDM 0
#endif

#if defined(__HIP_DEVICE_COMPILE__) && __has_builtin(__builtin_amdgcn_sched_barrier)
#define SCHED_FENCE() __builtin_amdgcn_sched_barrier(0)
#else
#define SCHED_FENCE()
#endif

__device__ __forceinline__ unsigned short f2bf(float f) {
    unsigned int u = __float_as_uint(f);
    u += 0x7FFFu + ((u >> 16) & 1u);   // round-to-nearest-even
    return (unsigned short)(u >> 16);
}

__device__ __forceinline__ float silu_f(float x) {
    return x * (1.0f / (1.0f + __expf(-x)));
}

#if HAVE_TDM
// Build a 2D tile descriptor (D#) and issue TENSOR_LOAD_TO_LDS.
// Tensor viewed from tile start; tdim1 = remaining rows (OOB rows read 0),
// tile = tile1 rows x tile0 elements, row stride = stride0 elements.
__device__ __forceinline__ void tdm_load_2d(const void* gptr, unsigned ldsOff,
                                            unsigned tdim0, unsigned tdim1,
                                            unsigned tile0, unsigned tile1,
                                            unsigned long long stride0) {
    unsigned long long ga = (unsigned long long)gptr;
    u32x4 g0;
    g0[0] = 1u;                                    // count=1, user mode
    g0[1] = ldsOff;                                // lds_addr (bytes)
    g0[2] = (unsigned)ga;                          // global_addr[31:0]
    g0[3] = (unsigned)((ga >> 32) & 0x01FFFFFFull) // global_addr[56:32]
          | (2u << 30);                            // type = 2 ("image")
    i32x8 g1;
    g1[0] = (int)(1u << 16);                       // wg_mask=0, data_size=1 (2B)
    g1[1] = (int)((tdim0 & 0xFFFFu) << 16);        // tensor_dim0[15:0]
    g1[2] = (int)((tdim0 >> 16) | ((tdim1 & 0xFFFFu) << 16)); // td0[31:16], td1[15:0]
    g1[3] = (int)((tdim1 >> 16) | (tile0 << 16));  // td1[31:16], tile_dim0
    g1[4] = (int)(tile1 & 0xFFFFu);                // tile_dim1 (tile_dim2 = 0)
    g1[5] = (int)(unsigned)(stride0 & 0xFFFFFFFFull);        // dim0_stride[31:0]
    g1[6] = (int)(unsigned)((stride0 >> 32) & 0xFFFFull);    // dim0_stride[47:32]
    g1[7] = 0;
    i32x4 z4; z4[0] = z4[1] = z4[2] = z4[3] = 0;
#if __clang_major__ >= 23
    i32x8 z8; z8[0]=z8[1]=z8[2]=z8[3]=z8[4]=z8[5]=z8[6]=z8[7]=0;
    __builtin_amdgcn_tensor_load_to_lds(g0, g1, z4, z4, z8, 0);
#else
    __builtin_amdgcn_tensor_load_to_lds(g0, g1, z4, z4, 0);
#endif
}
#endif

// ---------------------------------------------------------------------------
// f32 -> bf16 conversion (A-side matrices)
// ---------------------------------------------------------------------------
__global__ void cvt_f32_bf16_kernel(const float* __restrict__ in,
                                    unsigned short* __restrict__ out, int n) {
    int i = blockIdx.x * blockDim.x + threadIdx.x;
    if (i < n) out[i] = f2bf(in[i]);
}

// ---------------------------------------------------------------------------
// fused transpose + f32->bf16:  in[R][C] (f32) -> out[C][R] (bf16)
// ---------------------------------------------------------------------------
__global__ void transpose_cvt_kernel(const float* __restrict__ in,
                                     unsigned short* __restrict__ out,
                                     int R, int C) {
    __shared__ float tile[32][33];
    int c0 = blockIdx.x * 32, r0 = blockIdx.y * 32;
    int tx = threadIdx.x, ty = threadIdx.y;   // 32 x 8
    for (int i = ty; i < 32; i += 8) {
        int r = r0 + i, c = c0 + tx;
        tile[i][tx] = (r < R && c < C) ? in[(long long)r * C + c] : 0.0f;
    }
    __syncthreads();
    for (int i = ty; i < 32; i += 8) {
        int c = c0 + i, r = r0 + tx;
        if (c < C && r < R) out[(long long)c * R + r] = f2bf(tile[tx][i]);
    }
}

// ---------------------------------------------------------------------------
// Generic bf16 GEMM: C[M,N] = A[M,K] @ Bt[N,K]^T, f32 accumulate via WMMA.
// Block 256 threads = 8 waves (4M x 2N), block tile 64x256, wave tile 16x128.
// HBM->LDS staging by TDM (wave 0 issues descriptors, double buffered).
// mode 1: silu(acc+bias[n]) -> gates f32 / values^T bf16 / base f32 split
// mode 2: outF = acc + bias[n]
// mode 3: outBF = bf16(acc * aux[m,n])   (gate multiply)
// ---------------------------------------------------------------------------
#define GEMM_COMPUTE(LA, LB)                                                   \
    do {                                                                       \
        V16U a_;                                                               \
        a_.q[0] = *(const uint4*)(&(LA)[rowA * BK + kbA]);                     \
        a_.q[1] = *(const uint4*)(&(LA)[rowA * BK + kbA + 16]);                \
        V16U b_[8];                                                            \
        _Pragma("unroll")                                                      \
        for (int t = 0; t < 8; ++t) {                                          \
            int nloc = waveN * 128 + t * 16 + (lane & 15);                     \
            b_[t].q[0] = *(const uint4*)(&(LB)[nloc * BK + kbB]);              \
            b_[t].q[1] = *(const uint4*)(&(LB)[nloc * BK + kbB + 16]);         \
        }                                                                      \
        SCHED_FENCE(); /* keep all 18 ds_loads ahead of the WMMA chain */      \
        _Pragma("unroll")                                                      \
        for (int t = 0; t < 8; ++t)                                            \
            acc[t] = __builtin_amdgcn_wmma_f32_16x16x32_bf16(                  \
                false, a_.v, false, b_[t].v, (short)0, acc[t], false, false);  \
        SCHED_FENCE();                                                         \
    } while (0)

__global__ __launch_bounds__(256)
void gemm_bf16_kernel(const unsigned short* __restrict__ A,
                      const unsigned short* __restrict__ Bt,
                      int M, int N, int K,
                      long long sA, long long sB, long long sOut,
                      int mode,
                      const float* __restrict__ bias,
                      float* __restrict__ outF,
                      unsigned short* __restrict__ outBF,
                      const float* __restrict__ aux, long long sAux, int auxStride,
                      float* __restrict__ gatesOut,
                      unsigned short* __restrict__ valuesOut,
                      float* __restrict__ baseOut) {
    __shared__ alignas(32) unsigned short lA[2][BM * BK];   // 2 x 4 KB  [m][k]
    __shared__ alignas(32) unsigned short lB[2][BN * BK];   // 2 x 16 KB [n][k]

    const int tid   = threadIdx.x;
    const int wave  = tid >> 5;
    const int lane  = tid & 31;
    const int waveM = wave >> 1;
    const int waveN = wave & 1;
    const int m0 = blockIdx.y * BM;
    const int n0 = blockIdx.x * BN;
    const int bz = blockIdx.z;

    const unsigned short* Ab = A  + (long long)bz * sA;
    const unsigned short* Bb = Bt + (long long)bz * sB;

    v8f acc[8];
#pragma unroll
    for (int t = 0; t < 8; ++t)
#pragma unroll
        for (int r = 0; r < 8; ++r) acc[t][r] = 0.0f;

    const int rowA = waveM * 16 + (lane & 15);
    const int kbA  = (lane >> 4) * 8;    // A operand: high half-wave holds K+8
    const int kbB  = (lane >> 4) * 8;    // B operand: high half-wave holds K+8
    const int nIter = K / BK;

#if HAVE_TDM
    const unsigned offA0 = (unsigned)(uintptr_t)(&lA[0][0]);
    const unsigned offA1 = (unsigned)(uintptr_t)(&lA[1][0]);
    const unsigned offB0 = (unsigned)(uintptr_t)(&lB[0][0]);
    const unsigned offB1 = (unsigned)(uintptr_t)(&lB[1][0]);

    if (wave == 0) {
        tdm_load_2d(Ab + (long long)m0 * K, offA0,
                    (unsigned)K, (unsigned)(M - m0), BK, BM, (unsigned long long)K);
        tdm_load_2d(Bb + (long long)n0 * K, offB0,
                    (unsigned)K, (unsigned)(N - n0), BK, BN, (unsigned long long)K);
    }
    for (int it = 0; it < nIter; ++it) {
        const int cur = it & 1;
        if (wave == 0) {
            if (it + 1 < nIter) {       // prefetch next tiles into other buffer
                const int k1 = (it + 1) * BK;
                tdm_load_2d(Ab + (long long)m0 * K + k1, cur ? offA0 : offA1,
                            (unsigned)K, (unsigned)(M - m0), BK, BM,
                            (unsigned long long)K);
                tdm_load_2d(Bb + (long long)n0 * K + k1, cur ? offB0 : offB1,
                            (unsigned)K, (unsigned)(N - n0), BK, BN,
                            (unsigned long long)K);
                __builtin_amdgcn_s_wait_tensorcnt(2);   // current buffer done
            } else {
                __builtin_amdgcn_s_wait_tensorcnt(0);
            }
        }
        __syncthreads();
        GEMM_COMPUTE(lA[cur], lB[cur]);
        __syncthreads();
    }
#else
    for (int it = 0; it < nIter; ++it) {
        const int k0 = it * BK;
        // stage A: 64 rows x 32 k, 16B per chunk
        for (int c = tid; c < BM * 4; c += 256) {
            int r = c >> 2, s = (c & 3) * 8;
            uint4 v = make_uint4(0u, 0u, 0u, 0u);
            if (m0 + r < M) v = *(const uint4*)(Ab + (long long)(m0 + r) * K + k0 + s);
            *(uint4*)(&lA[0][r * BK + s]) = v;
        }
        // stage B^T: 256 rows x 32 k
        for (int c = tid; c < BN * 4; c += 256) {
            int r = c >> 2, s = (c & 3) * 8;
            uint4 v = make_uint4(0u, 0u, 0u, 0u);
            if (n0 + r < N) v = *(const uint4*)(Bb + (long long)(n0 + r) * K + k0 + s);
            *(uint4*)(&lB[0][r * BK + s]) = v;
        }
        __syncthreads();
        GEMM_COMPUTE(lA[0], lB[0]);
        __syncthreads();
    }
#endif

    // ---- epilogue (C layout: m = r + 8*(lane/16), n = lane%16 within tile)
#pragma unroll
    for (int t = 0; t < 8; ++t) {
        int n = n0 + waveN * 128 + t * 16 + (lane & 15);
        if (n >= N) continue;
#pragma unroll
        for (int r = 0; r < 8; ++r) {
            int m = m0 + waveM * 16 + r + 8 * (lane >> 4);
            if (m >= M) continue;
            float v = acc[t][r];
            if (mode == 1) {
                v = silu_f(v + bias[n]);
                if (n < PROJ) {
                    gatesOut[(long long)m * PROJ + n] = v;
                } else if (n < 2 * PROJ) {
                    // values stored TRANSPOSED per batch: valT[b][chan][tok]
                    int b = m >> 11, lm = m & (LL - 1), cn = n - PROJ;
                    valuesOut[(long long)b * PROJ * LL + (long long)cn * LL + lm] =
                        f2bf(v);
                } else {
                    baseOut[(long long)m * ATTN + (n - 2 * PROJ)] = v;
                }
            } else if (mode == 2) {
                outF[(long long)bz * sOut + (long long)m * N + n] = v + bias[n];
            } else if (mode == 3) {
                float g = aux[(long long)bz * sAux + (long long)m * auxStride + n];
                outBF[(long long)bz * sOut + (long long)m * N + n] = f2bf(v * g);
            } else {
                outF[(long long)bz * sOut + (long long)m * N + n] = v;
            }
        }
    }
}

// ---------------------------------------------------------------------------
// q/k generation: ms modulation + RoPE + q scaling, output bf16 [NTOK, 128]
// ---------------------------------------------------------------------------
__global__ void qk_rope_kernel(const float* __restrict__ base,
                               const float* __restrict__ msw,
                               const float* __restrict__ msb,
                               const float* __restrict__ scaling,
                               unsigned short* __restrict__ qout,
                               unsigned short* __restrict__ kout) {
    int tok = blockIdx.x;
    int d   = threadIdx.x;          // 0..63
    int l   = tok & (LL - 1);

    const float* bp = base + (long long)tok * ATTN;
    float x1 = bp[d], x2 = bp[d + 64];

    float q1 = x1 * msw[d]             + msb[d];
    float q2 = x2 * msw[d + 64]        + msb[d + 64];
    float k1 = x1 * msw[ATTN + d]      + msb[ATTN + d];
    float k2 = x2 * msw[ATTN + d + 64] + msb[ATTN + d + 64];

    float invf = __powf(10000.0f, -(float)d * (1.0f / 64.0f));
    float ang  = (float)l * invf;
    float s, c;
    __sincosf(ang, &s, &c);
    float sc = scaling[0];

    long long o = (long long)tok * ATTN;
    qout[o + d]      = f2bf((q1 * c - q2 * s) * sc);
    qout[o + d + 64] = f2bf((q2 * c + q1 * s) * sc);
    kout[o + d]      = f2bf(k1 * c - k2 * s);
    kout[o + d + 64] = f2bf(k2 * c + k1 * s);
}

// ---------------------------------------------------------------------------
// Attention logits + bias + softmax -> bf16 probabilities.
// Block = (b, 16-row tile); 16x2048 f32 logits in LDS (128 KB of 320 KB WGP LDS)
// ---------------------------------------------------------------------------
__global__ __launch_bounds__(256)
void attn_softmax_kernel(const unsigned short* __restrict__ qbf,
                         const unsigned short* __restrict__ kbf,
                         const float* __restrict__ bias,
                         unsigned short* __restrict__ attn) {
    __shared__ alignas(16) float slog[16 * LL];   // 128 KB

    const int b    = blockIdx.y;
    const int i0   = blockIdx.x * 16;
    const int tid  = threadIdx.x;
    const int wave = tid >> 5;
    const int lane = tid & 31;

    V16U qa[4];
    {
        int row = lane & 15;
        int kb  = (lane >> 4) * 8;
        const unsigned short* qrow = qbf + (long long)(b * LL + i0 + row) * ATTN;
#pragma unroll
        for (int c2 = 0; c2 < 4; ++c2) {
            qa[c2].q[0] = *(const uint4*)(qrow + c2 * 32 + kb);
            qa[c2].q[1] = *(const uint4*)(qrow + c2 * 32 + kb + 16);
        }
    }

    const int kbB = (lane >> 4) * 8;
    for (int jt = wave; jt < LL / 16; jt += 8) {
        v8f acc;
#pragma unroll
        for (int r = 0; r < 8; ++r) acc[r] = 0.0f;

        int jrow = jt * 16 + (lane & 15);
        const unsigned short* krow = kbf + (long long)(b * LL + jrow) * ATTN;
        // preload all 4 K-chunk operands, then chain 4 WMMAs
        V16U kb4[4];
#pragma unroll
        for (int c2 = 0; c2 < 4; ++c2) {
            kb4[c2].q[0] = *(const uint4*)(krow + c2 * 32 + kbB);
            kb4[c2].q[1] = *(const uint4*)(krow + c2 * 32 + kbB + 16);
        }
        SCHED_FENCE();
#pragma unroll
        for (int c2 = 0; c2 < 4; ++c2)
            acc = __builtin_amdgcn_wmma_f32_16x16x32_bf16(
                false, qa[c2].v, false, kb4[c2].v, (short)0, acc, false, false);
        SCHED_FENCE();

#pragma unroll
        for (int r = 0; r < 8; ++r) {
            int m = r + 8 * (lane >> 4);
            int i = i0 + m;
            int j = jt * 16 + (lane & 15);
            slog[m * LL + j] = acc[r] + bias[((long long)b * LL + i) * LL + j];
        }
    }
    __syncthreads();

    for (int rr = 0; rr < 2; ++rr) {
        int m = wave * 2 + rr;
        float mx = -3.0e38f;
        for (int j = lane; j < LL; j += 32) mx = fmaxf(mx, slog[m * LL + j]);
#pragma unroll
        for (int off = 16; off > 0; off >>= 1)
            mx = fmaxf(mx, __shfl_xor(mx, off, 32));
        float sum = 0.0f;
        for (int j = lane; j < LL; j += 32) {
            float e = __expf(slog[m * LL + j] - mx);
            slog[m * LL + j] = e;
            sum += e;
        }
#pragma unroll
        for (int off = 16; off > 0; off >>= 1)
            sum += __shfl_xor(sum, off, 32);
        float inv = 1.0f / sum;
        long long ob = ((long long)b * LL + i0 + m) * LL;
        for (int j = lane; j < LL; j += 32)
            attn[ob + j] = f2bf(slog[m * LL + j] * inv);
    }
}

// ---------------------------------------------------------------------------
// launch
// ---------------------------------------------------------------------------
extern "C" void kernel_launch(void* const* d_in, const int* in_sizes, int n_in,
                              void* d_out, int out_size, void* d_ws, size_t ws_size,
                              hipStream_t stream) {
    const float* node    = (const float*)d_in[0];
    const float* bias    = (const float*)d_in[1];
    const float* scaling = (const float*)d_in[2];
    const float* w1      = (const float*)d_in[3];
    const float* b1      = (const float*)d_in[4];
    const float* msw     = (const float*)d_in[5];
    const float* msb     = (const float*)d_in[6];
    const float* w2      = (const float*)d_in[7];
    const float* b2      = (const float*)d_in[8];
    float* out = (float*)d_out;
    char*  ws  = (char*)d_ws;

    // workspace layout (256B aligned)
    const size_t OFF_NODEBF = 0;
    const size_t OFF_W1T    = OFF_NODEBF + (size_t)NTOK * HID * 2;   // [NW1][HID]
    const size_t OFF_W2T    = OFF_W1T    + (size_t)HID * NW1 * 2;    // [HID][PROJ]
    const size_t OFF_GATES  = OFF_W2T    + (size_t)PROJ * HID * 2;
    const size_t OFF_VALT   = OFF_GATES  + (size_t)NTOK * PROJ * 4;  // [B][PROJ][LL]
    const size_t OFF_BASE   = OFF_VALT   + (size_t)NTOK * PROJ * 2;
    const size_t OFF_QBF    = OFF_BASE   + (size_t)NTOK * ATTN * 4;
    const size_t OFF_KBF    = OFF_QBF    + (size_t)NTOK * ATTN * 2;
    const size_t OFF_ATTNBF = OFF_KBF    + (size_t)NTOK * ATTN * 2;
    const size_t OFF_HBF    = OFF_ATTNBF + (size_t)BB * LL * (size_t)LL * 2;

    unsigned short* nodebf = (unsigned short*)(ws + OFF_NODEBF);
    unsigned short* w1T    = (unsigned short*)(ws + OFF_W1T);
    unsigned short* w2T    = (unsigned short*)(ws + OFF_W2T);
    float*          gates  = (float*)         (ws + OFF_GATES);
    unsigned short* valT   = (unsigned short*)(ws + OFF_VALT);
    float*          basef  = (float*)         (ws + OFF_BASE);
    unsigned short* qbf    = (unsigned short*)(ws + OFF_QBF);
    unsigned short* kbf    = (unsigned short*)(ws + OFF_KBF);
    unsigned short* attnbf = (unsigned short*)(ws + OFF_ATTNBF);
    unsigned short* hbf    = (unsigned short*)(ws + OFF_HBF);

    // 1) node -> bf16; w1 -> w1^T bf16 [NW1][HID]; w2 -> w2^T bf16 [HID][PROJ]
    {
        int n1 = NTOK * HID;
        cvt_f32_bf16_kernel<<<(n1 + 255) / 256, 256, 0, stream>>>(node, nodebf, n1);
        dim3 tb(32, 8);
        transpose_cvt_kernel<<<dim3((NW1 + 31) / 32, (HID + 31) / 32), tb, 0, stream>>>(
            w1, w1T, HID, NW1);
        transpose_cvt_kernel<<<dim3((HID + 31) / 32, (PROJ + 31) / 32), tb, 0, stream>>>(
            w2, w2T, PROJ, HID);
    }

    // 2) gemm1: silu(node@w1+b1) -> gates f32 / values^T bf16 / base f32
    {
        dim3 g((NW1 + BN - 1) / BN, NTOK / BM, 1);
        gemm_bf16_kernel<<<g, 256, 0, stream>>>(
            nodebf, w1T, NTOK, NW1, HID, 0, 0, 0,
            /*mode=*/1, b1, nullptr, nullptr, nullptr, 0, 0,
            gates, valT, basef);
    }

    // 3) q/k: ms modulation + rope + scaling
    qk_rope_kernel<<<NTOK, 64, 0, stream>>>(basef, msw, msb, scaling, qbf, kbf);

    // 4) logits + bias + softmax -> attn (bf16)
    {
        dim3 g(LL / 16, BB);
        attn_softmax_kernel<<<g, 256, 0, stream>>>(qbf, kbf, bias, attnbf);
    }

    // 5) h = (attn @ values) * gates  (bf16 out), batched over B
    {
        dim3 g(PROJ / BN, LL / BM, BB);
        gemm_bf16_kernel<<<g, 256, 0, stream>>>(
            attnbf, valT, LL, PROJ, LL,
            (long long)LL * LL, (long long)PROJ * LL, (long long)LL * PROJ,
            /*mode=*/3, nullptr, nullptr, hbf,
            gates, (long long)LL * PROJ, PROJ,
            nullptr, nullptr, nullptr);
    }

    // 6) out = h @ w2 + b2  (f32 out)
    {
        dim3 g(HID / BN, NTOK / BM, 1);
        gemm_bf16_kernel<<<g, 256, 0, stream>>>(
            hbf, w2T, NTOK, HID, PROJ, 0, 0, 0,
            /*mode=*/2, b2, out, nullptr, nullptr, 0, 0,
            nullptr, nullptr, nullptr);
    }
}
